// TGNRecommender_67714454389086
// MI455X (gfx1250) — matrix operand order repaired and checked by reading, not compiled, not run on hardware
//
#include <hip/hip_runtime.h>
#include <hip/hip_bf16.h>

typedef __attribute__((ext_vector_type(16))) _Float16 v16h;
typedef __attribute__((ext_vector_type(8)))  _Float16 v8h;
typedef __attribute__((ext_vector_type(8)))  float    v8f;

#define BATCH    262144
#define D        100
#define MSG      4
#define TD       100
#define KX       304      // 2*D + MSG + TD
#define KXP      320      // pad to mult of 32
#define KHP      128      // D padded
#define NG       300      // 3*D
#define NGP      304      // pad to mult of 16
#define BT       64       // batch rows per block
#define NBLK     (BATCH / BT)   // 4096

// ---------- LDS layout (overlaid phases), 215392 B total ----------
#define SX_OFF   0                         // _Float16 [64][320]  40960 B
#define SH_OFF   40960                     // _Float16 [64][128]  16384 B
#define SGI_OFF  57344                     // float    [64][304]  77824 B
#define SGH_OFF  135168                    // float    [64][304]  77824 B
#define SB_OFF   212992                    // float    [600] biases 2400 B
#define SHN_OFF  0                         // _Float16 [64][128]  (overlays sx)
#define SA_OFF   57344                     // float    [64][64]   (overlays sgi)
#define SMEM_SZ  215392

// ---------- workspace layout (bytes) ----------
#define WS_A      0ull                      // f16 [BATCH][64]   33554432
#define WS_PSUM   33554432ull               // f32 [4096][64]     1048576
#define WS_PSQ    34603008ull               // f32 [4096][64]     1048576
#define WS_STATS  35651584ull               // f32 [128] scale/shift
#define WS_TMP    35652096ull               // f32 [128] reduced sums
#define WS_WIH    35652608ull               // f16 [304][320]      194560
#define WS_WHH    35847168ull               // f16 [304][128]       77824
#define WS_W1     35924992ull               // f16 [64][128]        16384

// ---------------- kernel 0: convert weights to padded f16 ----------------
__global__ __launch_bounds__(256) void tgn_wconv(
    const float* __restrict__ W_ih, const float* __restrict__ W_hh,
    const float* __restrict__ W1f,
    _Float16* __restrict__ wih, _Float16* __restrict__ whh,
    _Float16* __restrict__ w1)
{
    int i = blockIdx.x * 256 + threadIdx.x;
    const int N0 = 304 * 320, N1 = 304 * 128, N2 = 64 * 128;
    if (i < N0) {
        int n = i / 320, k = i % 320;
        wih[i] = (_Float16)((n < NG && k < KX) ? W_ih[n * KX + k] : 0.0f);
    } else if (i < N0 + N1) {
        int j = i - N0; int n = j / 128, k = j % 128;
        whh[j] = (_Float16)((n < NG && k < D) ? W_hh[n * D + k] : 0.0f);
    } else if (i < N0 + N1 + N2) {
        int j = i - N0 - N1; int n = j / 128, k = j % 128;
        w1[j] = (_Float16)((k < D) ? W1f[n * D + k] : 0.0f);
    }
}

// ---------------- kernel 1: gather + GRU GEMMs + MLP1 (WMMA) ----------------
__global__ __launch_bounds__(256) void tgn_main(
    const int* __restrict__ n_id, const float* __restrict__ memory,
    const int* __restrict__ last_update, const int* __restrict__ store_src,
    const int* __restrict__ store_dst, const int* __restrict__ store_t,
    const float* __restrict__ store_msg, const float* __restrict__ time_w,
    const float* __restrict__ time_b, const float* __restrict__ b_ih,
    const float* __restrict__ b_hh, const float* __restrict__ b1,
    const _Float16* __restrict__ wih, const _Float16* __restrict__ whh,
    const _Float16* __restrict__ w1,
    _Float16* __restrict__ ws_a, float* __restrict__ ws_psum,
    float* __restrict__ ws_psq)
{
    __shared__ __align__(32) char smem[SMEM_SZ];
    _Float16* sx  = (_Float16*)(smem + SX_OFF);   // [64][KXP]
    _Float16* sh  = (_Float16*)(smem + SH_OFF);   // [64][KHP]
    float*    sgi = (float*)   (smem + SGI_OFF);  // [64][NGP]
    float*    sgh = (float*)   (smem + SGH_OFF);  // [64][NGP]
    float*    sb  = (float*)   (smem + SB_OFF);   // [600]: b_ih | b_hh
    _Float16* shn = (_Float16*)(smem + SHN_OFF);  // [64][KHP] overlay
    float*    sa  = (float*)   (smem + SA_OFF);   // [64][64]  overlay

    const int tid  = threadIdx.x;
    const int row0 = blockIdx.x * BT;

    // ---------- gather phase: 4 threads per batch row ----------
    {
        int r = tid >> 2, q = tid & 3;
        int n  = n_id[row0 + r];
        int s  = store_src[n];
        int d2 = store_dst[n];
        float dtv = (float)(store_t[n] - last_update[n]);
        const float* ms = memory + (long)s  * D;
        const float* md = memory + (long)d2 * D;
        const float* mh = memory + (long)n  * D;
        for (int k = q; k < D; k += 4) {
            sx[r * KXP + k]             = (_Float16)ms[k];
            sx[r * KXP + D + k]         = (_Float16)md[k];
            sx[r * KXP + 2*D + MSG + k] = (_Float16)__cosf(dtv * time_w[k] + time_b[k]);
            sh[r * KHP + k]             = (_Float16)mh[k];
        }
        if (q == 0) {
            const float* mm = store_msg + (long)n * MSG;
            for (int j = 0; j < MSG; ++j) sx[r * KXP + 2*D + j] = (_Float16)mm[j];
            for (int k = KX; k < KXP; ++k) sx[r * KXP + k] = (_Float16)0.0f;
            for (int k = D; k < KHP; ++k)  sh[r * KHP + k] = (_Float16)0.0f;
        }
        for (int e = tid; e < 600; e += 256)
            sb[e] = (e < 300) ? b_ih[e] : b_hh[e - 300];
    }
    __syncthreads();

    const int wid   = tid >> 5;
    const int lane  = tid & 31;
    const int nloc  = lane & 15;   // M index (A frag) / N index (B,C frags)
    const int khalf = lane >> 4;

    // ---------- GEMM1 + GEMM2, B-fragments held in registers across M tiles ----------
    for (int nt = wid; nt < 19; nt += 8) {
        const int ng = nt * 16 + nloc;
        // ---- GEMM1: gi = x @ W_ih^T  (K = 320) ----
        {
            v16h bfr[10];
            const _Float16* brow = wih + ng * KXP + khalf * 16;
            #pragma unroll
            for (int ks = 0; ks < 10; ++ks) bfr[ks] = *(const v16h*)(brow + ks * 32);
            #pragma unroll
            for (int mt = 0; mt < 4; ++mt) {
                v8f c = {};
                const _Float16* arow = sx + (mt * 16 + nloc) * KXP;
                #pragma unroll
                for (int ks = 0; ks < 10; ++ks) {
                    v16h a;
                    *((v8h*)&a)     = *(const v8h*)(arow + ks * 32 + khalf * 16);
                    *(((v8h*)&a)+1) = *(const v8h*)(arow + ks * 32 + khalf * 16 + 8);
                    c = __builtin_amdgcn_wmma_f32_16x16x32_f16(false, a, false, bfr[ks],
                                                               (short)0, c, false, false);
                }
                float* dst = sgi + (mt * 16 + khalf * 8) * NGP + nt * 16 + nloc;
                #pragma unroll
                for (int v = 0; v < 8; ++v) dst[v * NGP] = c[v];
            }
        }
        // ---- GEMM2: gh = h @ W_hh^T  (K = 128) ----
        {
            v16h bfr[4];
            const _Float16* brow = whh + ng * KHP + khalf * 16;
            #pragma unroll
            for (int ks = 0; ks < 4; ++ks) bfr[ks] = *(const v16h*)(brow + ks * 32);
            #pragma unroll
            for (int mt = 0; mt < 4; ++mt) {
                v8f c = {};
                const _Float16* arow = sh + (mt * 16 + nloc) * KHP;
                #pragma unroll
                for (int ks = 0; ks < 4; ++ks) {
                    v16h a;
                    *((v8h*)&a)     = *(const v8h*)(arow + ks * 32 + khalf * 16);
                    *(((v8h*)&a)+1) = *(const v8h*)(arow + ks * 32 + khalf * 16 + 8);
                    c = __builtin_amdgcn_wmma_f32_16x16x32_f16(false, a, false, bfr[ks],
                                                               (short)0, c, false, false);
                }
                float* dst = sgh + (mt * 16 + khalf * 8) * NGP + nt * 16 + nloc;
                #pragma unroll
                for (int v = 0; v < 8; ++v) dst[v * NGP] = c[v];
            }
        }
    }
    __syncthreads();

    // ---------- GRU elementwise: h_new -> shn (f16, overlays sx) ----------
    for (int e = tid; e < BT * D; e += 256) {
        int r = e / D, c = e % D;
        float ir  = sgi[r * NGP + c]       + sb[c];
        float iz  = sgi[r * NGP + D + c]   + sb[D + c];
        float in_ = sgi[r * NGP + 2*D + c] + sb[2*D + c];
        float hr  = sgh[r * NGP + c]       + sb[300 + c];
        float hz  = sgh[r * NGP + D + c]   + sb[300 + D + c];
        float hn  = sgh[r * NGP + 2*D + c] + sb[300 + 2*D + c];
        float rg = 1.0f / (1.0f + __expf(-(ir + hr)));
        float zg = 1.0f / (1.0f + __expf(-(iz + hz)));
        float ng = tanhf(in_ + rg * hn);
        float h  = (float)sh[r * KHP + c];
        shn[r * KHP + c] = (_Float16)((1.0f - zg) * ng + zg * h);
    }
    for (int e = tid; e < BT * (KHP - D); e += 256)
        shn[(e / (KHP - D)) * KHP + D + (e % (KHP - D))] = (_Float16)0.0f;
    __syncthreads();

    // ---------- GEMM3: a = relu(h_new @ W1^T + b1)  (M=64, N=64, K=128) ----------
    {
        const int nt = wid & 3, mtb = wid >> 2;   // waves sharing nt reuse B frags
        v16h bfr[4];
        const _Float16* brow = w1 + (nt * 16 + nloc) * KHP + khalf * 16;
        #pragma unroll
        for (int ks = 0; ks < 4; ++ks) bfr[ks] = *(const v16h*)(brow + ks * 32);
        float bb = b1[nt * 16 + nloc];
        #pragma unroll
        for (int mi = 0; mi < 2; ++mi) {
            const int mt = mtb + mi * 2;
            v8f c = {};
            const _Float16* arow = shn + (mt * 16 + nloc) * KHP;
            #pragma unroll
            for (int ks = 0; ks < 4; ++ks) {
                v16h a;
                *((v8h*)&a)     = *(const v8h*)(arow + ks * 32 + khalf * 16);
                *(((v8h*)&a)+1) = *(const v8h*)(arow + ks * 32 + khalf * 16 + 8);
                c = __builtin_amdgcn_wmma_f32_16x16x32_f16(false, a, false, bfr[ks],
                                                           (short)0, c, false, false);
            }
            float* dst = sa + (mt * 16 + khalf * 8) * 64 + nt * 16 + nloc;
            #pragma unroll
            for (int v = 0; v < 8; ++v) {
                float vv = c[v] + bb;
                dst[v * 64] = vv > 0.0f ? vv : 0.0f;
            }
        }
    }
    __syncthreads();

    // ---------- per-block BN partial sums (deterministic, no atomics) ----------
    if (tid < 64) {
        float s = 0.0f, q2 = 0.0f;
        for (int r = 0; r < BT; ++r) {
            float v = sa[r * 64 + tid];
            s += v; q2 += v * v;
        }
        ws_psum[blockIdx.x * 64 + tid] = s;
        ws_psq [blockIdx.x * 64 + tid] = q2;
    }
    // ---------- write activations (f16) ----------
    for (int i = 0; i < 16; ++i) {
        int e = tid + 256 * i;
        int r = e >> 6, c = e & 63;
        ws_a[(long)(row0 + r) * 64 + c] = (_Float16)sa[r * 64 + c];
    }
}

// ---------------- kernel 2a: reduce partials (128 blocks) ----------------
__global__ __launch_bounds__(256) void tgn_reduce(
    const float* __restrict__ psum, const float* __restrict__ psq,
    float* __restrict__ tmp)
{
    __shared__ float red[256];
    int s = blockIdx.x;               // 0..63 -> sum, 64..127 -> sumsq
    const float* src = (s < 64) ? psum : psq;
    int c = s & 63;
    float acc = 0.0f;
    for (int i = threadIdx.x; i < NBLK; i += 256) acc += src[i * 64 + c];
    red[threadIdx.x] = acc;
    __syncthreads();
    for (int off = 128; off > 0; off >>= 1) {
        if (threadIdx.x < off) red[threadIdx.x] += red[threadIdx.x + off];
        __syncthreads();
    }
    if (threadIdx.x == 0) tmp[s] = red[0];
}

// ---------------- kernel 2b: BN scale/shift ----------------
__global__ __launch_bounds__(64) void tgn_stats(
    const float* __restrict__ tmp, const float* __restrict__ gamma,
    const float* __restrict__ beta, float* __restrict__ stats)
{
    int c = threadIdx.x;
    if (c < 64) {
        const float invB = 1.0f / (float)BATCH;
        float mu  = tmp[c] * invB;
        float var = tmp[64 + c] * invB - mu * mu;
        float sc  = gamma[c] * rsqrtf(var + 1e-5f);
        stats[c]      = sc;
        stats[64 + c] = beta[c] - mu * sc;
    }
}

// ---------------- kernel 3: out = (a*scale+shift) @ W2^T + b2 ----------------
__global__ __launch_bounds__(256) void tgn_final(
    const _Float16* __restrict__ ws_a, const float* __restrict__ stats,
    const float* __restrict__ W2, const float* __restrict__ b2,
    float* __restrict__ out)
{
    __shared__ float sscale[64], sshift[64], sw2[128], sb2[2];
    int tid = threadIdx.x;
    if (tid < 64) { sscale[tid] = stats[tid]; sshift[tid] = stats[64 + tid]; }
    else if (tid < 192) sw2[tid - 64] = W2[tid - 64];
    else if (tid < 194) sb2[tid - 192] = b2[tid - 192];
    __syncthreads();
    long row = (long)blockIdx.x * 256 + tid;
    const v8h* ap = (const v8h*)(ws_a + row * 64);   // 8 x global_load_b128
    float a0 = sb2[0], a1 = sb2[1];
    #pragma unroll
    for (int j = 0; j < 8; ++j) {
        v8h av = ap[j];
        #pragma unroll
        for (int i = 0; i < 8; ++i) {
            int c = j * 8 + i;
            float an = (float)av[i] * sscale[c] + sshift[c];
            a0 += an * sw2[c];        // W2[0][c]
            a1 += an * sw2[64 + c];   // W2[1][c]
        }
    }
    out[row * 2]     = a0;
    out[row * 2 + 1] = a1;
}

// ---------------- host launcher ----------------
extern "C" void kernel_launch(void* const* d_in, const int* in_sizes, int n_in,
                              void* d_out, int out_size, void* d_ws, size_t ws_size,
                              hipStream_t stream)
{
    const int*   n_id        = (const int*)  d_in[0];
    const float* memory      = (const float*)d_in[1];
    const int*   last_update = (const int*)  d_in[2];
    const int*   store_src   = (const int*)  d_in[3];
    const int*   store_dst   = (const int*)  d_in[4];
    const int*   store_t     = (const int*)  d_in[5];
    const float* store_msg   = (const float*)d_in[6];
    const float* time_w      = (const float*)d_in[7];
    const float* time_b      = (const float*)d_in[8];
    const float* W_ih        = (const float*)d_in[9];
    const float* b_ih        = (const float*)d_in[10];
    const float* W_hh        = (const float*)d_in[11];
    const float* b_hh        = (const float*)d_in[12];
    const float* W1f         = (const float*)d_in[13];
    const float* b1          = (const float*)d_in[14];
    const float* gamma       = (const float*)d_in[15];
    const float* beta        = (const float*)d_in[16];
    const float* W2          = (const float*)d_in[17];
    const float* b2          = (const float*)d_in[18];
    float* out = (float*)d_out;

    char* ws = (char*)d_ws;
    _Float16* ws_a    = (_Float16*)(ws + WS_A);
    float*    ws_psum = (float*)   (ws + WS_PSUM);
    float*    ws_psq  = (float*)   (ws + WS_PSQ);
    float*    ws_stat = (float*)   (ws + WS_STATS);
    float*    ws_tmp  = (float*)   (ws + WS_TMP);
    _Float16* wih     = (_Float16*)(ws + WS_WIH);
    _Float16* whh     = (_Float16*)(ws + WS_WHH);
    _Float16* w1      = (_Float16*)(ws + WS_W1);

    const int wconv_elems = 304 * 320 + 304 * 128 + 64 * 128;
    tgn_wconv<<<(wconv_elems + 255) / 256, 256, 0, stream>>>(W_ih, W_hh, W1f, wih, whh, w1);

    tgn_main<<<NBLK, 256, 0, stream>>>(n_id, memory, last_update, store_src,
                                       store_dst, store_t, store_msg, time_w,
                                       time_b, b_ih, b_hh, b1, wih, whh, w1,
                                       ws_a, ws_psum, ws_psq);

    tgn_reduce<<<128, 256, 0, stream>>>(ws_psum, ws_psq, ws_tmp);
    tgn_stats<<<1, 64, 0, stream>>>(ws_tmp, gamma, beta, ws_stat);
    tgn_final<<<BATCH / 256, 256, 0, stream>>>(ws_a, ws_stat, W2, b2, out);
}
